// Layer_19868518711857
// MI455X (gfx1250) — matrix operand order repaired and checked by reading, not compiled, not run on hardware
//
#include <hip/hip_runtime.h>
#include <hip/hip_bf16.h>

// ---------------- problem constants ----------------
#define BATCH      4
#define CPD        64
#define RES        64
#define NSAMP      131072
#define TFRAMES    512
#define FS         2048
#define HID        16

// conv tiling
#define OUT_CHUNK  4096                    // outputs per workgroup
#define WIN        (OUT_CHUNK + FS + 32)   // 6176 energy-window elems (mult of 16)
#define FRP        (FS + 64)               // 2112 padded reversed filter elems
#define KSTEPS     65                      // K = 2080 = 65 * 32
#define TILES_PER_WAVE 2                   // 16 tiles / 8 waves

typedef __attribute__((ext_vector_type(16))) __bf16 v16bf;
typedef __attribute__((ext_vector_type(8)))  float  v8f;

// async-DMA pointer types (match builtin signature: V4i AS1 / AS3)
typedef int v4i __attribute__((vector_size(4 * sizeof(int))));
typedef __attribute__((address_space(1))) v4i gv4i;
typedef __attribute__((address_space(3))) v4i lv4i;

union ABu {
    unsigned int u[8];
    uint4        q[2];
    v16bf        v;
};

__device__ __forceinline__ unsigned short f2bf(float x) {
    unsigned int b = __float_as_uint(x);
    b += 0x7FFFu + ((b >> 16) & 1u);       // round-to-nearest-even
    return (unsigned short)(b >> 16);
}

// padded reversed filter: frp[i] = f[2079 - i] for 32 <= i <= 2079, else 0
__device__ __forceinline__ unsigned short frp_elem(const float* __restrict__ frow, int i) {
    if (i < 32 || i > 2079) return (unsigned short)0;
    return f2bf(frow[2079 - i]);
}

// ---------------- K1: sequential damping scan ----------------
// out[0] = f[0]; out[t] = (f[t] + out[t-1]) * d[t]
__global__ __launch_bounds__(32)
void scan_kernel(const float* __restrict__ forces,
                 const float* __restrict__ dampmod,
                 const float* __restrict__ damppar,
                 float* __restrict__ damped) {
    int gid = blockIdx.x * 32 + threadIdx.x;      // 0..255 = b*64 + c
    int c   = gid & 63;
    const float* frow = forces  + gid * TFRAMES;
    const float* mrow = dampmod + gid * TFRAMES;
    float*       orow = damped  + gid * TFRAMES;

    float sg    = 1.0f / (1.0f + __expf(-damppar[c]));
    float dbase = 0.5f + sg * 0.4999f;            // BASE + sig * (MAX-BASE)

    float carry = 0.0f;
    for (int t = 0; t < TFRAMES; ++t) {
        float d = dbase - fabsf(mrow[t]);
        d = fminf(fmaxf(d, 0.0f), 1.0f);
        float o = (t == 0) ? frow[0] : (frow[t] + carry) * d;
        orow[t] = o;
        carry = o;
    }
}

// ---------------- K2: hypernetwork + routing einsum ----------------
// routed[b,d,t] = sum_c x[c] * (routing[c,d] + b2[c,d] + sum_j h[j] W2[j,c,d])
__global__ __launch_bounds__(64)
void hyper_kernel(const float* __restrict__ damped,
                  const float* __restrict__ routing,
                  const float* __restrict__ W1,
                  const float* __restrict__ b1,
                  const float* __restrict__ W2,
                  const float* __restrict__ b2,
                  float* __restrict__ routed) {
    __shared__ float sx[CPD];
    __shared__ float sh[HID];
    int t = blockIdx.x, b = blockIdx.y, tid = threadIdx.x;

    sx[tid] = damped[(b * CPD + tid) * TFRAMES + t];
    __syncthreads();

    if (tid < HID) {
        float a = b1[tid];
        #pragma unroll 8
        for (int c = 0; c < CPD; ++c) a = fmaf(sx[c], W1[c * HID + tid], a);
        sh[tid] = (a > 0.0f) ? a : 0.2f * a;      // leaky_relu(0.2)
    }
    __syncthreads();

    int d = tid;
    float acc = 0.0f;
    for (int c = 0; c < CPD; ++c) {
        float w = routing[c * RES + d] + b2[c * RES + d];
        const float* w2p = W2 + c * RES + d;
        #pragma unroll
        for (int j = 0; j < HID; ++j)
            w = fmaf(sh[j], w2p[j * (CPD * RES)], w);
        acc = fmaf(sx[c], w, acc);
    }
    routed[(b * RES + d) * TFRAMES + t] = acc;
}

// ---------------- K3: to_ctrl GEMM ----------------
// to_ctrl[b,c,t] = sum_r routed[b,r,t] * to_control[r,c]
__global__ __launch_bounds__(256)
void toctrl_kernel(const float* __restrict__ routed,
                   const float* __restrict__ tc,
                   float* __restrict__ out) {
    int idx = blockIdx.x * 256 + threadIdx.x;     // = (b*64+c)*512 + t
    int t = idx & (TFRAMES - 1);
    int c = (idx >> 9) & 63;
    int b = idx >> 15;
    float acc = 0.0f;
    #pragma unroll 8
    for (int r = 0; r < RES; ++r)
        acc = fmaf(routed[(b * RES + r) * TFRAMES + t], tc[r * CPD + c], acc);
    out[idx] = acc;
}

// ---------------- K4: upsample * noise + 2048-tap FIR via bf16 WMMA ----------------
// out[n] = sum_{k=0}^{2047} e[n-k] f[k]   ==  C[m,j] = sum_c E[m,c] F[c,j]
//   E[m,c] = e[n0 + 16m + c]   (contiguous energy rows, stride 16)
//   F[c,j] = f[j - c]          (Toeplitz filter matrix, data-independent)
// Noise window staged via async global->LDS DMA (ASYNCcnt), overlapped with
// filter-table construction.
__global__ __launch_bounds__(256)
void conv_wmma_kernel(const float* __restrict__ noise,
                      const float* __restrict__ filters,
                      const float* __restrict__ routed,
                      float* __restrict__ outRes) {
    __shared__ alignas(16) float          sN[WIN];        // raw f32 noise window (DMA)
    __shared__ alignas(16) unsigned short sE[WIN];        // bf16 energy window
    __shared__ unsigned int sFrE[FRP / 2];                // packed pairs, even start
    __shared__ unsigned int sFrO[FRP / 2];                // packed pairs, odd start
    __shared__ float sR[TFRAMES];                         // routed row (control rate)

    const int chunk   = blockIdx.x;          // 0..31
    const int r       = blockIdx.y;          // 0..63
    const int b       = blockIdx.z;          // 0..3
    const int ch      = b * RES + r;
    const int s       = chunk * OUT_CHUNK;   // chunk output start
    const int winBase = s - FS;              // first energy sample in window
    const int tid     = threadIdx.x;

    const float* nrow = noise + (size_t)ch * NSAMP;

    // ---- issue async DMA: raw noise window -> LDS (16B granules) ----
    // Granules are 16B aligned and NSAMP is a multiple of 4, so no granule
    // straddles the buffer ends; out-of-range granules clamp to a valid
    // address (their data is discarded by the bounds check below).
    for (int i = 4 * tid; i < WIN; i += 4 * 256) {
        int n  = winBase + i;
        int gn = n < 0 ? 0 : (n > NSAMP - 4 ? NSAMP - 4 : n);
#if __has_builtin(__builtin_amdgcn_global_load_async_to_lds_b128)
        __builtin_amdgcn_global_load_async_to_lds_b128(
            (gv4i*)(nrow + gn), (lv4i*)(&sN[i]), 0, 0);
#else
        unsigned lofs = (unsigned)(size_t)(&sN[i]);      // low 32b = LDS offset
        asm volatile("global_load_async_to_lds_b128 %0, %1, off"
                     :: "v"(lofs), "v"(nrow + gn) : "memory");
#endif
    }

    // ---- overlap with DMA: control row + Toeplitz filter tables ----
    for (int i = tid; i < TFRAMES; i += 256)
        sR[i] = routed[ch * TFRAMES + i];

    const float* frow = filters + r * FS;
    for (int k = tid; k < FRP / 2; k += 256) {
        unsigned short f0  = frp_elem(frow, 2 * k);
        unsigned short f1  = frp_elem(frow, 2 * k + 1);
        unsigned short f2v = frp_elem(frow, 2 * k + 2);
        sFrE[k] = (unsigned)f0 | ((unsigned)f1 << 16);
        sFrO[k] = (unsigned)f1 | ((unsigned)f2v << 16);
    }

    // ---- drain DMA, then sync the workgroup ----
#if __has_builtin(__builtin_amdgcn_s_wait_asynccnt)
    __builtin_amdgcn_s_wait_asynccnt(0);
#else
    asm volatile("s_wait_asynccnt 0x0" ::: "memory");
#endif
    __syncthreads();

    // ---- bf16 energy window: linear-interp upsample * noise ----
    for (int i = tid; i < WIN; i += 256) {
        int n = winBase + i;
        float e = 0.0f;
        if (n >= 0 && n < NSAMP) {
            float pos = (n + 0.5f) * (1.0f / 256.0f) - 0.5f;
            pos = fminf(fmaxf(pos, 0.0f), (float)(TFRAMES - 1));
            int   i0 = (int)pos;                       // floor (pos >= 0)
            int   i1 = min(i0 + 1, TFRAMES - 1);
            float fr = pos - (float)i0;
            float up = sR[i0] * (1.0f - fr) + sR[i1] * fr;
            e = up * sN[i];
        }
        sE[i] = f2bf(e);
    }
    __syncthreads();

    // ---- WMMA main loop ----
    const int wave = tid >> 5;
    const int lane = tid & 31;
    const int hh   = lane >> 4;       // lane half (K-half select)
    const int jm   = lane & 15;       // A row m == B/C column j

    v8f acc[TILES_PER_WAVE] = {};
    const int tb0 = wave * TILES_PER_WAVE * 256;   // first tile base, rel. to s

    for (int ks = 0; ks < KSTEPS; ++ks) {
        // B fragment: F[c,j] = f[j-c] from reversed filter; 8 aligned b32 loads
        int e0p = 32 * ks + 16 * hh - jm + 31;     // in [16, 2095]
        const unsigned int* tbl = (e0p & 1) ? sFrO : sFrE;
        int half = e0p >> 1;
        ABu bu;
        #pragma unroll
        for (int v = 0; v < 8; ++v) bu.u[v] = tbl[half + v];

        #pragma unroll
        for (int t = 0; t < TILES_PER_WAVE; ++t) {
            // A fragment: contiguous energy row, 2 x ds_load_b128 per lane
            int rowStart = tb0 + t * 256 + 16 * jm + 32 * ks;  // multiple of 16
            ABu au;
            au.q[0] = *reinterpret_cast<const uint4*>(&sE[rowStart + 16 * hh]);
            au.q[1] = *reinterpret_cast<const uint4*>(&sE[rowStart + 16 * hh + 8]);
            acc[t] = __builtin_amdgcn_wmma_f32_16x16x32_bf16(
                false, au.v, false, bu.v, (short)0, acc[t], false, false);
        }
    }

    // ---- store: C layout — lane half hh, col jm, VGPR g -> row m = g + 8*hh
    float* orow = outRes + (size_t)ch * NSAMP;
    #pragma unroll
    for (int t = 0; t < TILES_PER_WAVE; ++t) {
        int base = s + tb0 + t * 256;
        #pragma unroll
        for (int g = 0; g < 8; ++g) {
            int samp = base + 16 * (g + 8 * hh) + jm;
            orow[samp] = acc[t][g];
        }
    }
}

// ---------------- launch ----------------
extern "C" void kernel_launch(void* const* d_in, const int* in_sizes, int n_in,
                              void* d_out, int out_size, void* d_ws, size_t ws_size,
                              hipStream_t stream) {
    (void)in_sizes; (void)n_in; (void)out_size; (void)ws_size;

    const float* forces     = (const float*)d_in[0];
    const float* dampmod    = (const float*)d_in[1];
    const float* noise      = (const float*)d_in[2];
    const float* damppar    = (const float*)d_in[3];
    const float* routing    = (const float*)d_in[4];
    const float* W1         = (const float*)d_in[5];
    const float* b1         = (const float*)d_in[6];
    const float* W2         = (const float*)d_in[7];
    const float* b2         = (const float*)d_in[8];
    const float* filters    = (const float*)d_in[9];
    const float* to_control = (const float*)d_in[10];

    float* out_ctrl = (float*)d_out;                         // (4,64,512)
    float* out_res  = (float*)d_out + BATCH * CPD * TFRAMES; // (4,64,131072)

    float* damped = (float*)d_ws;                            // 131072 f32
    float* routed = damped + BATCH * CPD * TFRAMES;          // 131072 f32

    scan_kernel  <<<dim3(8),              dim3(32),  0, stream>>>(forces, dampmod, damppar, damped);
    hyper_kernel <<<dim3(TFRAMES, BATCH), dim3(64),  0, stream>>>(damped, routing, W1, b1, W2, b2, routed);
    toctrl_kernel<<<dim3(512),            dim3(256), 0, stream>>>(routed, to_control, out_ctrl);
    conv_wmma_kernel<<<dim3(NSAMP / OUT_CHUNK, RES, BATCH), dim3(256), 0, stream>>>(
        noise, filters, routed, out_res);
}
